// HAN_62105227100409
// MI455X (gfx1250) — matrix operand order repaired and checked by reading, not compiled, not run on hardware
//
#include <hip/hip_runtime.h>
#include <hip/hip_bf16.h>

// ---------------- problem constants (match reference) ----------------
#define N_NODES 20000
#define E_EDGES 320000
#define M_PATHS 3
#define IN_DIM  256
#define HID     64
#define NH1     4
#define F1      (NH1*HID)   // 256
#define OUT_DIM 256
#define NHO     1
#define SA_H    128
#define NEG_SLOPE 0.2f

typedef __attribute__((ext_vector_type(16))) _Float16 v16h;
typedef __attribute__((ext_vector_type(8)))  float    v8f;
typedef __attribute__((ext_vector_type(4)))  int      v4i;

#ifndef __has_builtin
#define __has_builtin(x) 0
#endif
#if defined(__HIP_DEVICE_COMPILE__) && \
    __has_builtin(__builtin_amdgcn_global_load_async_to_lds_b128) && \
    __has_builtin(__builtin_amdgcn_s_wait_asynccnt)
#define USE_ASYNC_LDS 1
// pointer flavors the builtin expects (from hipcc diagnostic):
typedef __attribute__((address_space(1))) v4i* gv4i_p;   // global (device) v4i*
typedef __attribute__((address_space(3))) v4i* lv4i_p;   // LDS v4i*
#else
#define USE_ASYNC_LDS 0
#endif

static inline int ceil_div_l(long a, long b) { return (int)((a + b - 1) / b); }

// ------- monotonic float<->uint encoding for atomic float max -------
__device__ __forceinline__ unsigned f32_ord(float f) {
  unsigned u = __float_as_uint(f);
  return (u & 0x80000000u) ? ~u : (u | 0x80000000u);
}
__device__ __forceinline__ float f32_unord(unsigned u) {
  return (u & 0x80000000u) ? __uint_as_float(u ^ 0x80000000u)
                           : __uint_as_float(~u);
}

// ---------------- conversion kernels ----------------
__global__ void cvt_f32_to_f16(const float* __restrict__ in,
                               _Float16* __restrict__ out, long n) {
  long i = (long)blockIdx.x * blockDim.x + threadIdx.x;
  if (i < n) out[i] = (_Float16)in[i];
}

// in: [rows, cols] row-major f32 -> out: [cols, rows] row-major f16 (transpose)
__global__ void cvt_f32_to_f16_T(const float* __restrict__ in,
                                 _Float16* __restrict__ out, int rows, int cols) {
  long i = (long)blockIdx.x * blockDim.x + threadIdx.x;
  long n = (long)rows * cols;
  if (i >= n) return;
  int r = (int)(i / cols), c = (int)(i % cols);
  out[(long)c * rows + r] = (_Float16)in[i];
}

// ---------------- WMMA GEMM: C[Md,Nc] = A[Md,K](f16) @ Bt[Nc,K]^T (f16) --------
// Block = 8 waves = 8 M-tiles sharing ONE N-tile. The 16xK strip of Bt for that
// N-tile is staged in LDS once per block (async-to-LDS when available), then
// each wave runs the K-loop with A fragments from global (b128, coalesced) and
// B fragments from LDS (ds_load_b128; rows padded 16B for bank spread).
// Fragment layouts per CDNA5 ISA 7.12.2:
//   A 16x32 f16 : lane L holds row m=L%16; kbase=(L<16?0:8);
//                 half-pair j<4 -> K=kbase+2j,+1 ; j>=4 -> K=kbase+16+2(j-4),+1
//   B 32x16 f16 : lane L holds col n=L%16; same K order as A (kbase=(L<16?0:8))
//   C 16x16 f32 : VGPR r -> row r + 8*(L>=16), col n=L%16
#define LDS_PAD 8                     // halves (16B) of per-row padding
__global__ void wmma_gemm_f16(const _Float16* __restrict__ A,
                              const _Float16* __restrict__ Bt,
                              float* __restrict__ C,
                              int Md, int Nc, int K) {
  __shared__ _Float16 lbs[16 * (256 + LDS_PAD)];   // K <= 256
  const int ntile = Nc >> 4;
  const int mblk  = blockIdx.x / ntile;
  const int nt    = blockIdx.x - mblk * ntile;
  const int wid   = threadIdx.x >> 5;
  const int lane  = threadIdx.x & 31;
  const int ldl   = K + LDS_PAD;                   // LDS row stride in halves
  const int cpr   = K >> 3;                        // 16B chunks per row

  // ---- stage B strip: 16 rows of Bt, full K ----
  const _Float16* gb = Bt + (size_t)(nt * 16) * K;
  const int nvec = 16 * cpr;
  for (int i = threadIdx.x; i < nvec; i += blockDim.x) {
    int r = i / cpr, c = i - r * cpr;
#if USE_ASYNC_LDS
    __builtin_amdgcn_global_load_async_to_lds_b128(
        (gv4i_p)(gb + (size_t)r * K + c * 8),
        (lv4i_p)(lbs + r * ldl + c * 8), 0, 0);
#else
    *(uint4*)(lbs + r * ldl + c * 8) =
        *(const uint4*)(gb + (size_t)r * K + c * 8);
#endif
  }
#if USE_ASYNC_LDS
  __builtin_amdgcn_s_wait_asynccnt(0);
#endif
  __syncthreads();

  const int mt = mblk * 8 + wid;
  if (mt * 16 >= Md) return;

  const int m = mt * 16 + (lane & 15);
  const int kbaseA = (lane >> 4) * 8;    // A fragment K base for this lane
  const int kbaseB = (lane >> 4) * 16;   // unused; B fragment uses kbaseA

  const _Float16* arow = A + (size_t)m * K;
  const _Float16* brow = lbs + (lane & 15) * ldl;

  v8f acc = {};
  for (int k0 = 0; k0 < K; k0 += 32) {
    __builtin_prefetch(arow + k0 + 64, 0, 1);   // global_prefetch_b8
    v16h a, b;
#pragma unroll
    for (int j = 0; j < 8; ++j) {
      int ka = k0 + kbaseA + ((j < 4) ? (2 * j) : (16 + 2 * (j - 4)));
      int kb = k0 + kbaseA + ((j < 4) ? (2 * j) : (16 + 2 * (j - 4)));
      a[2 * j]     = arow[ka];
      a[2 * j + 1] = arow[ka + 1];
      b[2 * j]     = brow[kb];
      b[2 * j + 1] = brow[kb + 1];
    }
    acc = __builtin_amdgcn_wmma_f32_16x16x32_f16(false, a, false, b,
                                                 (short)0, acc, false, false);
  }
  const int n     = nt * 16 + (lane & 15);
  const int rbase = mt * 16 + (lane >> 4) * 8;
#pragma unroll
  for (int r = 0; r < 8; ++r)
    C[(size_t)(rbase + r) * Nc + n] = acc[r];
}

// ---------------- GAT attention pieces ----------------
// el/er: one thread per (node, head); 64..256-length dot products.
__global__ void compute_elr(const float* __restrict__ feat,
                            const float* __restrict__ al,
                            const float* __restrict__ ar,
                            float* __restrict__ el, float* __restrict__ er,
                            int Nn, int NHk, int Dk) {
  long i = (long)blockIdx.x * blockDim.x + threadIdx.x;
  if (i >= (long)Nn * NHk) return;
  int n = (int)(i / NHk), h = (int)(i % NHk);
  const float* f = feat + (size_t)n * NHk * Dk + (size_t)h * Dk;
  const float* L = al + (size_t)h * Dk;
  const float* R = ar + (size_t)h * Dk;
  float sl = 0.f, sr = 0.f;
  for (int j = 0; j < Dk; ++j) { float v = f[j]; sl += v * L[j]; sr += v * R[j]; }
  el[i] = sl; er[i] = sr;
}

// pass 1: per-edge leaky-relu score, atomic max into emax[dst] (ordered-uint)
__global__ void edge_max(const int* __restrict__ src, const int* __restrict__ dst,
                         const float* __restrict__ el, const float* __restrict__ er,
                         unsigned* __restrict__ emax, int Ek, int NHk) {
  long i = (long)blockIdx.x * blockDim.x + threadIdx.x;
  if (i >= (long)Ek * NHk) return;
  int e = (int)(i / NHk), h = (int)(i % NHk);
  int s = src[e], d = dst[e];
  float x = el[(size_t)s * NHk + h] + er[(size_t)d * NHk + h];
  x = (x > 0.f) ? x : NEG_SLOPE * x;
  atomicMax(&emax[(size_t)d * NHk + h], f32_ord(x));
}

// pass 2 (fused): ee = exp(e - max); denom[dst] += ee; z[dst] += ee * feat[src]
// One wave32 per edge: lane c covers feature channels c, c+32, ... (8 dwords/lane).
// Dk = 1<<dshift.
__global__ void edge_accum(const int* __restrict__ src, const int* __restrict__ dst,
                           const float* __restrict__ el, const float* __restrict__ er,
                           const unsigned* __restrict__ emax,
                           const float* __restrict__ feat,
                           float* __restrict__ denom, float* __restrict__ zout,
                           int Ek, int NHk, int dshift, int ldz) {
  int wave = (int)((blockIdx.x * blockDim.x + threadIdx.x) >> 5);
  int lane = threadIdx.x & 31;
  if (wave >= Ek) return;
  int s = src[wave], d = dst[wave];
  int F = NHk << dshift;
  float ee[4];
  for (int h = 0; h < NHk; ++h) {
    float x = el[(size_t)s * NHk + h] + er[(size_t)d * NHk + h];
    x = (x > 0.f) ? x : NEG_SLOPE * x;
    ee[h] = __expf(x - f32_unord(emax[(size_t)d * NHk + h]));
  }
  if (lane < NHk) atomicAdd(&denom[(size_t)d * NHk + lane], ee[lane]);
  const float* frow = feat + (size_t)s * F;
  float* zrow = zout + (size_t)d * ldz;
  for (int c = lane; c < F; c += 32)
    atomicAdd(&zrow[c], ee[c >> dshift] * frow[c]);
}

// per-node: z /= (denom + 1e-9), optional relu; Dk = 1<<dshift
__global__ void node_norm(float* __restrict__ zout, const float* __restrict__ denom,
                          int Nn, int NHk, int dshift, int ldz, int do_relu) {
  long i = (long)blockIdx.x * blockDim.x + threadIdx.x;
  int F = NHk << dshift;
  if (i >= (long)Nn * F) return;
  int n = (int)(i / F), c = (int)(i % F);
  float v = zout[(size_t)n * ldz + c] /
            (denom[(size_t)n * NHk + (c >> dshift)] + 1e-9f);
  if (do_relu) v = (v > 0.f) ? v : 0.f;
  zout[(size_t)n * ldz + c] = v;
}

// ---------------- semantic attention ----------------
// per row (n*M+m) of t[rows,SA_H]: score = sum_j tanh(t+b1) * v; wave reduce;
// atomicAdd into wsum[m]
__global__ void sa_score(const float* __restrict__ t, const float* __restrict__ b1,
                         const float* __restrict__ v, float* __restrict__ wsum,
                         int rows, int SAHk, int Mk) {
  int row  = (int)((blockIdx.x * blockDim.x + threadIdx.x) >> 5);
  int lane = threadIdx.x & 31;
  if (row >= rows) return;
  float s = 0.f;
  const float* tr = t + (size_t)row * SAHk;
  for (int j = lane; j < SAHk; j += 32) s += tanhf(tr[j] + b1[j]) * v[j];
#pragma unroll
  for (int off = 16; off > 0; off >>= 1) s += __shfl_down(s, off, 32);
  if (lane == 0) atomicAdd(&wsum[row % Mk], s);
}

// tiny: beta = softmax(wsum / N); optionally copy beta to att_out (att_mp)
__global__ void sa_beta(const float* __restrict__ wsum, float* __restrict__ beta,
                        float* __restrict__ att_out, int Mk, float invN) {
  if (blockIdx.x != 0 || threadIdx.x != 0) return;
  float w[8], mx = -3.0e38f;
  for (int m = 0; m < Mk; ++m) { w[m] = wsum[m] * invN; mx = fmaxf(mx, w[m]); }
  float sum = 0.f;
  for (int m = 0; m < Mk; ++m) { w[m] = __expf(w[m] - mx); sum += w[m]; }
  for (int m = 0; m < Mk; ++m) {
    float b = w[m] / sum;
    beta[m] = b;
    if (att_out) att_out[m] = b;
  }
}

// out[n,f] = sum_m beta[m] * z[n,m,f]
__global__ void sa_combine(const float* __restrict__ z, const float* __restrict__ beta,
                           float* __restrict__ out, int Nn, int F, int Mk) {
  long i = (long)blockIdx.x * blockDim.x + threadIdx.x;
  if (i >= (long)Nn * F) return;
  int n = (int)(i / F), c = (int)(i % F);
  float s = 0.f;
  for (int m = 0; m < Mk; ++m)
    s += beta[m] * z[((size_t)n * Mk + m) * F + c];
  out[i] = s;
}

// ---------------- driver ----------------
extern "C" void kernel_launch(void* const* d_in, const int* in_sizes, int n_in,
                              void* d_out, int out_size, void* d_ws, size_t ws_size,
                              hipStream_t stream) {
  const float* x     = (const float*)d_in[0];
  const int*   edges = (const int*)  d_in[1];   // [M, 2, E]
  const float* W1    = (const float*)d_in[2];   // [256, 256]
  const float* al1   = (const float*)d_in[3];   // [4, 64]
  const float* ar1   = (const float*)d_in[4];
  const float* saW1  = (const float*)d_in[5];   // [256, 128]
  const float* sab1  = (const float*)d_in[6];   // [128]
  const float* sav1  = (const float*)d_in[7];   // [128, 1]
  const float* W2    = (const float*)d_in[8];   // [256, 256]
  const float* al2   = (const float*)d_in[9];   // [1, 256]
  const float* ar2   = (const float*)d_in[10];
  const float* saW2  = (const float*)d_in[11];  // [256, 128]
  const float* sab2  = (const float*)d_in[12];
  const float* sav2  = (const float*)d_in[13];
  float* out = (float*)d_out;                   // [N*256] h  +  [3] att_mp

  // bump allocator over d_ws
  char* wsb = (char*)d_ws;
  size_t off = 0;
  auto alloc_ = [&](size_t bytes) -> void* {
    void* p = wsb + off;
    off = (off + bytes + 255) & ~(size_t)255;
    return p;
  };
#define ALLOC(T, cnt) (T*)alloc_((size_t)(cnt) * sizeof(T))

  float*    zbuf  = ALLOC(float,    (size_t)N_NODES * M_PATHS * F1);  // z (both layers)
  float*    featb = ALLOC(float,    (size_t)N_NODES * F1);            // feat (both layers)
  float*    hbuf  = ALLOC(float,    (size_t)N_NODES * F1);            // SA1 output
  float*    thid  = ALLOC(float,    (size_t)N_NODES * M_PATHS * SA_H);// SA hidden
  _Float16* habuf = ALLOC(_Float16, (size_t)N_NODES * M_PATHS * F1);  // f16 activations (reused)
  _Float16* W1t   = ALLOC(_Float16, (size_t)F1 * IN_DIM);
  _Float16* W2t   = ALLOC(_Float16, (size_t)OUT_DIM * F1);
  _Float16* saW1t = ALLOC(_Float16, (size_t)SA_H * F1);
  _Float16* saW2t = ALLOC(_Float16, (size_t)SA_H * OUT_DIM);
  float*    el    = ALLOC(float,    (size_t)N_NODES * NH1);
  float*    er    = ALLOC(float,    (size_t)N_NODES * NH1);
  unsigned* emax  = ALLOC(unsigned, (size_t)N_NODES * NH1);
  float*    denom = ALLOC(float,    (size_t)N_NODES * NH1);
  float*    wsum  = ALLOC(float,    16);
  float*    beta  = ALLOC(float,    16);

  const int BLK = 256;
  auto g1 = [&](long n) { return dim3((unsigned)ceil_div_l(n, BLK)); };
  // block = 8 M-tiles x 1 N-tile
  auto gemm_grid = [&](int Md, int Nc) {
    long mblks = (Md / 16 + 7) / 8;
    return dim3((unsigned)(mblks * (Nc / 16)));
  };

  // ---- weight conversion (transposed to [Nc, K] f16) ----
  cvt_f32_to_f16_T<<<g1((long)IN_DIM * F1), BLK, 0, stream>>>(W1,   W1t,   IN_DIM, F1);
  cvt_f32_to_f16_T<<<g1((long)F1 * OUT_DIM), BLK, 0, stream>>>(W2,  W2t,   F1, OUT_DIM);
  cvt_f32_to_f16_T<<<g1((long)F1 * SA_H), BLK, 0, stream>>>(saW1,  saW1t, F1, SA_H);
  cvt_f32_to_f16_T<<<g1((long)OUT_DIM * SA_H), BLK, 0, stream>>>(saW2, saW2t, OUT_DIM, SA_H);

  const long NF  = (long)N_NODES * F1;          // 5.12M
  const long NMF = (long)N_NODES * M_PATHS * F1;
  const long SAr = (long)N_NODES * M_PATHS;     // 60000 rows

  // ================= Layer 1 =================
  cvt_f32_to_f16<<<g1(NF), BLK, 0, stream>>>(x, habuf, NF);
  wmma_gemm_f16<<<gemm_grid(N_NODES, F1), BLK, 0, stream>>>(habuf, W1t, featb,
                                                            N_NODES, F1, IN_DIM);
  compute_elr<<<g1((long)N_NODES * NH1), BLK, 0, stream>>>(featb, al1, ar1, el, er,
                                                           N_NODES, NH1, HID);
  (void)hipMemsetAsync(zbuf, 0, (size_t)NMF * sizeof(float), stream);
  for (int m = 0; m < M_PATHS; ++m) {
    const int* src = edges + (size_t)m * 2 * E_EDGES;
    const int* dst = src + E_EDGES;
    (void)hipMemsetAsync(emax,  0, (size_t)N_NODES * NH1 * sizeof(unsigned), stream);
    (void)hipMemsetAsync(denom, 0, (size_t)N_NODES * NH1 * sizeof(float), stream);
    edge_max<<<g1((long)E_EDGES * NH1), BLK, 0, stream>>>(src, dst, el, er, emax,
                                                          E_EDGES, NH1);
    edge_accum<<<g1((long)E_EDGES * 32), BLK, 0, stream>>>(src, dst, el, er, emax,
                                                           featb, denom,
                                                           zbuf + (size_t)m * F1,
                                                           E_EDGES, NH1, /*dshift=*/6,
                                                           M_PATHS * F1);
    node_norm<<<g1(NF), BLK, 0, stream>>>(zbuf + (size_t)m * F1, denom,
                                          N_NODES, NH1, /*dshift=*/6, M_PATHS * F1, 1);
  }
  // semantic attention 1
  cvt_f32_to_f16<<<g1(NMF), BLK, 0, stream>>>(zbuf, habuf, NMF);
  wmma_gemm_f16<<<gemm_grid((int)SAr, SA_H), BLK, 0, stream>>>(habuf, saW1t, thid,
                                                               (int)SAr, SA_H, F1);
  (void)hipMemsetAsync(wsum, 0, 16 * sizeof(float), stream);
  sa_score<<<g1(SAr * 32), BLK, 0, stream>>>(thid, sab1, sav1, wsum,
                                             (int)SAr, SA_H, M_PATHS);
  sa_beta<<<1, 32, 0, stream>>>(wsum, beta, nullptr, M_PATHS, 1.0f / N_NODES);
  sa_combine<<<g1(NF), BLK, 0, stream>>>(zbuf, beta, hbuf, N_NODES, F1, M_PATHS);

  // ================= Layer 2 =================
  cvt_f32_to_f16<<<g1(NF), BLK, 0, stream>>>(hbuf, habuf, NF);
  wmma_gemm_f16<<<gemm_grid(N_NODES, OUT_DIM), BLK, 0, stream>>>(habuf, W2t, featb,
                                                                 N_NODES, OUT_DIM, F1);
  compute_elr<<<g1((long)N_NODES * NHO), BLK, 0, stream>>>(featb, al2, ar2, el, er,
                                                           N_NODES, NHO, OUT_DIM);
  (void)hipMemsetAsync(zbuf, 0, (size_t)NMF * sizeof(float), stream);
  for (int m = 0; m < M_PATHS; ++m) {
    const int* src = edges + (size_t)m * 2 * E_EDGES;
    const int* dst = src + E_EDGES;
    (void)hipMemsetAsync(emax,  0, (size_t)N_NODES * NHO * sizeof(unsigned), stream);
    (void)hipMemsetAsync(denom, 0, (size_t)N_NODES * NHO * sizeof(float), stream);
    edge_max<<<g1((long)E_EDGES * NHO), BLK, 0, stream>>>(src, dst, el, er, emax,
                                                          E_EDGES, NHO);
    edge_accum<<<g1((long)E_EDGES * 32), BLK, 0, stream>>>(src, dst, el, er, emax,
                                                           featb, denom,
                                                           zbuf + (size_t)m * OUT_DIM,
                                                           E_EDGES, NHO, /*dshift=*/8,
                                                           M_PATHS * OUT_DIM);
    node_norm<<<g1(NF), BLK, 0, stream>>>(zbuf + (size_t)m * OUT_DIM, denom,
                                          N_NODES, NHO, /*dshift=*/8,
                                          M_PATHS * OUT_DIM, 0);
  }
  // semantic attention 2 (final): h -> d_out, beta -> att_mp at d_out + N*256
  cvt_f32_to_f16<<<g1(NMF), BLK, 0, stream>>>(zbuf, habuf, NMF);
  wmma_gemm_f16<<<gemm_grid((int)SAr, SA_H), BLK, 0, stream>>>(habuf, saW2t, thid,
                                                               (int)SAr, SA_H, OUT_DIM);
  (void)hipMemsetAsync(wsum, 0, 16 * sizeof(float), stream);
  sa_score<<<g1(SAr * 32), BLK, 0, stream>>>(thid, sab2, sav2, wsum,
                                             (int)SAr, SA_H, M_PATHS);
  sa_beta<<<1, 32, 0, stream>>>(wsum, beta, out + (size_t)N_NODES * OUT_DIM,
                                M_PATHS, 1.0f / N_NODES);
  sa_combine<<<g1(NF), BLK, 0, stream>>>(zbuf, beta, out, N_NODES, OUT_DIM, M_PATHS);
#undef ALLOC
}
